// SegBOW_37160057045658
// MI455X (gfx1250) — matrix operand order repaired and checked by reading, not compiled, not run on hardware
//
#include <hip/hip_runtime.h>
#include <hip/hip_bf16.h>

typedef __attribute__((ext_vector_type(16))) _Float16 v16h;
typedef __attribute__((ext_vector_type(8)))  _Float16 v8h;
typedef __attribute__((ext_vector_type(8)))  float    v8f;

#define NB 16
#define NS 32
#define NT 512
#define NV 32000
#define ND 128
#define KCHUNK 32
#define LPAD 40   // halfs per LDS row: 80B => 16B-aligned frags, conflict-free banks

__global__ __launch_bounds__(256) void segbow_wmma_kernel(
    const int*  __restrict__ tokens,   // [B,T]
    const int*  __restrict__ lengths,  // [B]
    const int*  __restrict__ spans,    // [B,S,2]
    const float* __restrict__ W,       // [V,D]
    const float* __restrict__ bias,    // [D]
    float* __restrict__ out)           // [B,S,D]
{
    __shared__ __attribute__((aligned(16))) _Float16 EbufT[ND][LPAD]; // [d][kk] (transposed E chunk)
    __shared__ __attribute__((aligned(16))) _Float16 Abuf[NS][LPAD];  // [s][kk] (0/1 mask chunk)
    __shared__ int toks[NT];
    __shared__ int iarr[NS];
    __shared__ int jarr[NS];

    const int b    = blockIdx.x;
    const int tid  = threadIdx.x;
    const int wave = tid >> 5;        // 0..7 -> N tile (16 cols of D)
    const int lane = tid & 31;

    // Stage this batch's tokens into LDS once.
    for (int t = tid; t < NT; t += 256) toks[t] = tokens[b * NT + t];
    // Clamp span boundaries to the sample length.
    if (tid < NS) {
        int len = lengths[b];
        int i0  = spans[(b * NS + tid) * 2 + 0];
        int j0  = spans[(b * NS + tid) * 2 + 1];
        iarr[tid] = i0 < len ? i0 : len;
        jarr[tid] = j0 < len ? j0 : len;
    }
    __syncthreads();

    v8f acc0 = {};  // M-tile 0: spans 0..15
    v8f acc1 = {};  // M-tile 1: spans 16..31

    // Gather assignment: 8 threads per token row, 16 consecutive d each.
    const int grow = tid >> 3;            // 0..31 token within chunk
    const int gcol = (tid & 7) * 16;      // starting d
    // Mask-build assignment: 8 threads per span, 4 k each.
    const int arow = tid >> 3;            // span s
    const int ak0  = (tid & 7) * 4;

    const int l16  = lane & 15;
    const int kofs = (lane >> 4) << 3;    // A frag K offset: 0 or 8
    const int kb   = (lane >> 4) << 4;    // unused (B frags use kofs): 0 or 16
    const int col  = wave * 16 + l16;     // this lane's D column

    for (int kc = 0; kc < NT / KCHUNK; ++kc) {
        // ---- Gather E chunk: E[kk][d] = W[tok[kc*32+kk]][d], stored transposed as f16
        {
            int tok = toks[kc * KCHUNK + grow];
            const float* wr = W + (long)tok * ND + gcol;
            #pragma unroll
            for (int q = 0; q < 16; ++q)
                EbufT[gcol + q][grow] = (_Float16)wr[q];
        }
        // ---- Build 0/1 mask chunk: Abuf[s][kk] = (kc*32+kk) in [i_s, j_s)
        {
            int i0 = iarr[arow], j0 = jarr[arow];
            #pragma unroll
            for (int q = 0; q < 4; ++q) {
                int t = kc * KCHUNK + ak0 + q;
                Abuf[arow][ak0 + q] = (t >= i0 && t < j0) ? (_Float16)1.0f : (_Float16)0.0f;
            }
        }
        __syncthreads();

        // ---- Load fragments per documented CDNA5 wave32 layouts
        union { v16h v; v8h h[2]; } a0, a1, bf;
        // A (16x32 f16): lanes 0-15 row=l16, K = {kofs..kofs+7, 16+kofs..16+kofs+7}
        a0.h[0] = *(const v8h*)&Abuf[l16][kofs];
        a0.h[1] = *(const v8h*)&Abuf[l16][16 + kofs];
        a1.h[0] = *(const v8h*)&Abuf[16 + l16][kofs];
        a1.h[1] = *(const v8h*)&Abuf[16 + l16][16 + kofs];
        // B (32x16 f16): lane holds column `col`, K = {kofs..kofs+7, 16+kofs..16+kofs+7}
        bf.h[0] = *(const v8h*)&EbufT[col][kofs];
        bf.h[1] = *(const v8h*)&EbufT[col][16 + kofs];

        acc0 = __builtin_amdgcn_wmma_f32_16x16x32_f16(false, a0.v, false, bf.v,
                                                      (short)0, acc0, false, false);
        acc1 = __builtin_amdgcn_wmma_f32_16x16x32_f16(false, a1.v, false, bf.v,
                                                      (short)0, acc1, false, false);
        __syncthreads();
    }

    // ---- Store: f32 C/D layout — VGPR r: lanes 0-15 M=r, lanes 16-31 M=8+r; N=l16
    const int mb   = (lane >> 4) * 8;
    const float bb = bias[col];
    #pragma unroll
    for (int r = 0; r < 8; ++r) {
        out[(b * NS + (mb + r)) * ND + col]        = acc0[r] + bb;
        out[(b * NS + 16 + (mb + r)) * ND + col]   = acc1[r] + bb;
    }
}

extern "C" void kernel_launch(void* const* d_in, const int* in_sizes, int n_in,
                              void* d_out, int out_size, void* d_ws, size_t ws_size,
                              hipStream_t stream) {
    const int*   tokens  = (const int*)d_in[0];
    const int*   lengths = (const int*)d_in[1];
    const int*   spans   = (const int*)d_in[2];
    const float* W       = (const float*)d_in[3];
    const float* bias    = (const float*)d_in[4];
    float*       out     = (float*)d_out;

    segbow_wmma_kernel<<<NB, 256, 0, stream>>>(tokens, lengths, spans, W, bias, out);
}
